// MultiHeadAttention_31009663877223
// MI455X (gfx1250) — matrix operand order repaired and checked by reading, not compiled, not run on hardware
//
#include <hip/hip_runtime.h>

// ---------------- CDNA5 WMMA types ----------------
typedef __attribute__((ext_vector_type(16))) __bf16 v16bf;
typedef __attribute__((ext_vector_type(8)))  float  v8f;

__device__ __forceinline__ v8f wmma_bf16(v16bf a, v16bf b, v8f c) {
  // D = A(16x32 bf16) * B(32x16 bf16) + C(16x16 f32)
  return __builtin_amdgcn_wmma_f32_16x16x32_bf16(
      /*neg_a=*/false, a, /*neg_b=*/false, b,
      /*c_mod=*/(short)0, c, /*reuse_a=*/false, /*reuse_b=*/false);
}

// ---- CDNA5 async global->LDS copy (ASYNCcnt path, no VGPR round trip) ----
// Flat pointers to __shared__ carry the LDS byte offset in their low 32 bits
// (ISA 10.2: LDS aperture maps addr[31:0] -> LDS_ADDR).
__device__ __forceinline__ void async_copy_b128(void* lds_dst, const void* gsrc) {
  unsigned lds = (unsigned)(unsigned long long)lds_dst;
  asm volatile("global_load_async_to_lds_b128 %0, %1, off"
               :: "v"(lds), "v"(gsrc)
               : "memory");
}
__device__ __forceinline__ void wait_async0() {
  asm volatile("s_wait_asynccnt 0x0" ::: "memory");
}

// A-fragment (16x32, bf16), row-major with row stride ld (ISA 7.12.2):
// lanes 0-15: M=0..15, K={0..7,16..23}; lanes 16-31: M=0..15, K={8..15,24..31}.
__device__ __forceinline__ v16bf load_a_frag(const __bf16* p, int ld) {
  const int lane = threadIdx.x & 31;
  const __bf16* row = p + (size_t)(lane & 15) * ld + ((lane & 16) ? 8 : 0);
  v16bf r;
#pragma unroll
  for (int i = 0; i < 8; ++i) r[i] = row[i];
#pragma unroll
  for (int i = 0; i < 8; ++i) r[i + 8] = row[16 + i];
  return r;
}

// B-fragment (32x16) from a TRANSPOSED tile Bt[n][k] with row stride ld:
// n = lane&15; lanes 0-15 take K={0..7,16..23}, lanes 16-31 take K={8..15,24..31}.
// Two 16B runs per lane -> 2x ds_load_b128.
__device__ __forceinline__ v16bf load_bT_frag(const __bf16* p, int ld) {
  const int lane = threadIdx.x & 31;
  const __bf16* row = p + (size_t)(lane & 15) * ld + ((lane & 16) ? 8 : 0);
  v16bf r;
#pragma unroll
  for (int i = 0; i < 8; ++i) r[i] = row[i];
#pragma unroll
  for (int i = 0; i < 8; ++i) r[i + 8] = row[16 + i];
  return r;
}

// 16-lane-group reductions (rows of a C tile live in one 16-lane half)
__device__ __forceinline__ float rowmax16(float v) {
#pragma unroll
  for (int m = 8; m >= 1; m >>= 1) v = fmaxf(v, __shfl_xor(v, m, 32));
  return v;
}
__device__ __forceinline__ float rowsum16(float v) {
#pragma unroll
  for (int m = 8; m >= 1; m >>= 1) v += __shfl_xor(v, m, 32);
  return v;
}

// ---------------- fp32 -> bf16 convert ----------------
__global__ void f32_to_bf16_kernel(const float* __restrict__ in,
                                   __bf16* __restrict__ out, int n) {
  int i = blockIdx.x * blockDim.x + threadIdx.x;
  if (i < n) out[i] = (__bf16)in[i];
}

// ---------------- bf16 WMMA GEMM ----------------
// C[M,N] = A[M,K] @ B[K,N].  Block tile 128x128, 8 waves (4 in M x 2 in N),
// each wave computes 32x64 via 2x4 WMMA tiles; K-step 32 = one WMMA.
// A tile staged via async global->LDS; B tile staged transposed (Bt[n][k]).
template <bool OUT_F32>
__global__ __launch_bounds__(256)
void gemm_bf16_wmma(const __bf16* __restrict__ A, const __bf16* __restrict__ B,
                    __bf16* __restrict__ Cb, float* __restrict__ Cf,
                    const float* __restrict__ bias, int M, int N, int K) {
  __shared__ __bf16 As[128][40];  // [m][k], 32+8 pad, row stride 80B (16B mult)
  __shared__ __bf16 Bt[128][40];  // [n][k] transposed, row stride 80B

  const int tid = threadIdx.x;
  const int lane = tid & 31;
  const int wave = tid >> 5;
  const int wm = wave & 3;   // 0..3 -> M (32 rows each)
  const int wn = wave >> 2;  // 0..1 -> N (64 cols each)
  const int m0 = blockIdx.y * 128;
  const int n0 = blockIdx.x * 128;

  v8f acc[2][4] = {};

  for (int kk = 0; kk < K; kk += 32) {
    // ---- A tile 128x32: async copy straight into LDS (2 x b128/thread) ----
#pragma unroll
    for (int it = 0; it < 2; ++it) {
      int l = (tid + it * 256) * 8;
      int r = l >> 5, c = l & 31;
      async_copy_b128(&As[r][c], &A[(size_t)(m0 + r) * K + kk + c]);
    }
    // ---- B tile 32x128: load b128, store transposed into Bt[n][k] ----
#pragma unroll
    for (int it = 0; it < 2; ++it) {
      int l = (tid + it * 256) * 8;
      int r = l >> 7, c = l & 127;  // r = k row (0..31), c = n col
      uint4 t4 = *(const uint4*)&B[(size_t)(kk + r) * N + n0 + c];
      const __bf16* e = (const __bf16*)&t4;
#pragma unroll
      for (int j = 0; j < 8; ++j) Bt[c + j][r] = e[j];
    }
    if (kk + 32 < K) {  // prefetch next K tile (global_prefetch_b8)
      __builtin_prefetch(&A[(size_t)(m0 + (tid & 127)) * K + kk + 32], 0, 1);
      __builtin_prefetch(&B[(size_t)(kk + 32 + (tid & 31)) * N + n0], 0, 1);
    }
    wait_async0();
    __syncthreads();

    v16bf a0 = load_a_frag(&As[wm * 32][0], 40);
    v16bf a1 = load_a_frag(&As[wm * 32 + 16][0], 40);
#pragma unroll
    for (int j = 0; j < 4; ++j) {
      v16bf bj = load_bT_frag(&Bt[wn * 64 + j * 16][0], 40);
      acc[0][j] = wmma_bf16(a0, bj, acc[0][j]);
      acc[1][j] = wmma_bf16(a1, bj, acc[1][j]);
    }
    __syncthreads();
  }

  // C/D layout: lane holds column n = lane&15; VGPR r -> row r (+8 lanes>=16)
  const int ncol = lane & 15;
  const int mhalf = (lane & 16) ? 8 : 0;
#pragma unroll
  for (int i = 0; i < 2; ++i)
#pragma unroll
    for (int j = 0; j < 4; ++j)
#pragma unroll
      for (int r = 0; r < 8; ++r) {
        int m = m0 + wm * 32 + i * 16 + mhalf + r;
        int n = n0 + wn * 64 + j * 16 + ncol;
        float v = acc[i][j][r];
        if (OUT_F32)
          Cf[(size_t)m * N + n] = v + bias[n];
        else
          Cb[(size_t)m * N + n] = (__bf16)v;
      }
}

// ---------------- Flash attention (causal) ----------------
// Q,K,V,Ctx: bf16 [B, S, H*64] row-major.  Block: 128 threads (4 waves),
// 64 query rows per block (16 per wave); streams 64-key K/V blocks via LDS.
// K staged async (row-major, QK^T B-frags read rows contiguously);
// V staged TRANSPOSED (Vt[hd][key]) so PV B-frags read rows contiguously.
__global__ __launch_bounds__(128)
void flash_attn_wmma(const __bf16* __restrict__ Qg, const __bf16* __restrict__ Kg,
                     const __bf16* __restrict__ Vg, __bf16* __restrict__ Ctx,
                     int S, int Hn, float scale) {
  __shared__ __bf16 Ks[64][72];     // [key][hd]
  __shared__ __bf16 Vt[64][72];     // [hd][key] (transposed V)
  __shared__ __bf16 Ps[4][16][72];  // per-wave P staging (C-layout -> A-layout)

  const int tid = threadIdx.x;
  const int lane = tid & 31;
  const int w = tid >> 5;
  const int qb = blockIdx.x;
  const int bh = blockIdx.y;
  const int b = bh / Hn, h = bh % Hn;
  const int q0 = qb * 64;
  const int DM = Hn * 64;
  const size_t base = (size_t)b * S * DM + (size_t)h * 64;

  // Q fragments for this wave's 16 rows, HD=64 -> two 16x32 A-frags
  v16bf qf[2];
  {
    const __bf16* qrow =
        Qg + base + (size_t)(q0 + w * 16 + (lane & 15)) * DM + ((lane & 16) ? 8 : 0);
#pragma unroll
    for (int c = 0; c < 2; ++c) {
#pragma unroll
      for (int i = 0; i < 8; ++i) qf[c][i] = qrow[c * 32 + i];
#pragma unroll
      for (int i = 0; i < 8; ++i) qf[c][8 + i] = qrow[c * 32 + 16 + i];
    }
  }

  v8f o[4] = {};  // ctx accum 16x64 (4 hd-tiles)
  float mrow[8], lrow[8];
#pragma unroll
  for (int r = 0; r < 8; ++r) { mrow[r] = -1e30f; lrow[r] = 0.f; }

  const int half = (lane & 16) ? 8 : 0;
  const int n16 = lane & 15;

  for (int kb = 0; kb <= qb; ++kb) {
    __syncthreads();  // previous iteration's readers done (WAR on Ks/Vt)
    // ---- K block 64x64: async global->LDS (4 x b128/thread) ----
#pragma unroll
    for (int it = 0; it < 4; ++it) {
      int l = (tid + it * 128) * 8;
      int r = l >> 6, c = l & 63;
      async_copy_b128(&Ks[r][c], &Kg[base + (size_t)(kb * 64 + r) * DM + c]);
    }
    // ---- V block 64x64: load b128, store transposed Vt[hd][key] ----
#pragma unroll
    for (int it = 0; it < 4; ++it) {
      int l = (tid + it * 128) * 8;
      int key = l >> 6, c = l & 63;  // c = hd
      uint4 t4 = *(const uint4*)&Vg[base + (size_t)(kb * 64 + key) * DM + c];
      const __bf16* e = (const __bf16*)&t4;
#pragma unroll
      for (int j = 0; j < 8; ++j) Vt[c + j][key] = e[j];
    }
    wait_async0();
    __syncthreads();

    // ---- scores: S = Q(16x64) @ K^T(64x64) as 4 tiles of 16 keys ----
    float sc[4][8];
    const bool diag = (kb == qb);
#pragma unroll
    for (int t = 0; t < 4; ++t) {
      v8f s = {};
#pragma unroll
      for (int c = 0; c < 2; ++c) {
        // B(kk=hd, n=key) = Ks[key][hd]: contiguous read along Ks row
        s = wmma_bf16(qf[c], load_bT_frag(&Ks[t * 16][c * 32], 72), s);
      }
#pragma unroll
      for (int r = 0; r < 8; ++r) {
        float v = s[r] * scale;
        if (diag) {
          int kglob = kb * 64 + t * 16 + n16;
          int qglob = q0 + w * 16 + half + r;
          if (kglob > qglob) v = -1e30f;
        }
        sc[t][r] = v;
      }
    }

    // ---- online softmax per row ----
#pragma unroll
    for (int r = 0; r < 8; ++r) {
      float mx = fmaxf(fmaxf(sc[0][r], sc[1][r]), fmaxf(sc[2][r], sc[3][r]));
      mx = rowmax16(mx);
      float mnew = fmaxf(mrow[r], mx);
      float alpha = __expf(mrow[r] - mnew);
      float ps = 0.f;
#pragma unroll
      for (int t = 0; t < 4; ++t) {
        float p = __expf(sc[t][r] - mnew);
        sc[t][r] = p;
        ps += p;
      }
      ps = rowsum16(ps);
      lrow[r] = lrow[r] * alpha + ps;
      mrow[r] = mnew;
#pragma unroll
      for (int t = 0; t < 4; ++t) o[t][r] *= alpha;
    }

    // ---- P (C-layout) -> LDS -> A-layout fragments ----
#pragma unroll
    for (int t = 0; t < 4; ++t)
#pragma unroll
      for (int r = 0; r < 8; ++r)
        Ps[w][half + r][t * 16 + n16] = (__bf16)sc[t][r];
    __syncthreads();

    // ---- O += P(16x64) @ V(64x64) ----
#pragma unroll
    for (int kc = 0; kc < 2; ++kc) {
      v16bf pf = load_a_frag(&Ps[w][0][kc * 32], 72);
#pragma unroll
      for (int t = 0; t < 4; ++t) {
        // B(kk=key, n=hd) = Vt[hd][key]: contiguous read along Vt row
        o[t] = wmma_bf16(pf, load_bT_frag(&Vt[t * 16][kc * 32], 72), o[t]);
      }
    }
  }

  // ---- epilogue: normalize, write ctx as [b, q, h*64 + hd] bf16 ----
#pragma unroll
  for (int r = 0; r < 8; ++r) {
    float inv = 1.0f / lrow[r];
    size_t q = (size_t)(q0 + w * 16 + half + r);
#pragma unroll
    for (int t = 0; t < 4; ++t)
      Ctx[base + q * DM + t * 16 + n16] = (__bf16)(o[t][r] * inv);
  }
}

// ---------------- host launcher ----------------
extern "C" void kernel_launch(void* const* d_in, const int* in_sizes, int n_in,
                              void* d_out, int out_size, void* d_ws, size_t ws_size,
                              hipStream_t stream) {
  (void)in_sizes; (void)n_in; (void)out_size; (void)ws_size;

  const int Bn = 2, S = 2048, DM = 1024, Hn = 16;
  const size_t MT = (size_t)Bn * S;  // 4096 rows

  const float* x  = (const float*)d_in[0];
  const float* Wq = (const float*)d_in[1];
  const float* Wk = (const float*)d_in[2];
  const float* Wv = (const float*)d_in[3];
  const float* Wo = (const float*)d_in[4];
  const float* bo = (const float*)d_in[5];
  float* out = (float*)d_out;

  // workspace carve-up (bf16 buffers, all 16B-aligned slices)
  char* ws = (char*)d_ws;
  __bf16* xb  = (__bf16*)ws; ws += MT * DM * sizeof(__bf16);
  __bf16* wqb = (__bf16*)ws; ws += (size_t)DM * DM * sizeof(__bf16);
  __bf16* wkb = (__bf16*)ws; ws += (size_t)DM * DM * sizeof(__bf16);
  __bf16* wvb = (__bf16*)ws; ws += (size_t)DM * DM * sizeof(__bf16);
  __bf16* wob = (__bf16*)ws; ws += (size_t)DM * DM * sizeof(__bf16);
  __bf16* Qb  = (__bf16*)ws; ws += MT * DM * sizeof(__bf16);
  __bf16* Kb  = (__bf16*)ws; ws += MT * DM * sizeof(__bf16);
  __bf16* Vb  = (__bf16*)ws; ws += MT * DM * sizeof(__bf16);
  __bf16* Cxb = (__bf16*)ws; ws += MT * DM * sizeof(__bf16);

  // fp32 -> bf16 conversions
  {
    int n = (int)(MT * DM);
    f32_to_bf16_kernel<<<(n + 255) / 256, 256, 0, stream>>>(x, xb, n);
    int nw = DM * DM;
    f32_to_bf16_kernel<<<(nw + 255) / 256, 256, 0, stream>>>(Wq, wqb, nw);
    f32_to_bf16_kernel<<<(nw + 255) / 256, 256, 0, stream>>>(Wk, wkb, nw);
    f32_to_bf16_kernel<<<(nw + 255) / 256, 256, 0, stream>>>(Wv, wvb, nw);
    f32_to_bf16_kernel<<<(nw + 255) / 256, 256, 0, stream>>>(Wo, wob, nw);
  }

  // Q/K/V projections: [4096,1024] @ [1024,1024] -> bf16
  dim3 gg(DM / 128, (unsigned)(MT / 128));
  gemm_bf16_wmma<false><<<gg, 256, 0, stream>>>(xb, wqb, Qb, nullptr, nullptr,
                                                (int)MT, DM, DM);
  gemm_bf16_wmma<false><<<gg, 256, 0, stream>>>(xb, wkb, Kb, nullptr, nullptr,
                                                (int)MT, DM, DM);
  gemm_bf16_wmma<false><<<gg, 256, 0, stream>>>(xb, wvb, Vb, nullptr, nullptr,
                                                (int)MT, DM, DM);

  // causal flash attention, scale = 1/sqrt(64)
  dim3 ga(S / 64, Bn * Hn);
  flash_attn_wmma<<<ga, 128, 0, stream>>>(Qb, Kb, Vb, Cxb, S, Hn, 0.125f);

  // output projection: ctx @ Wo + bo -> fp32 d_out
  gemm_bf16_wmma<true><<<gg, 256, 0, stream>>>(Cxb, wob, nullptr, out, bo,
                                               (int)MT, DM, DM);
}
